// MesoNet_62646392979829
// MI455X (gfx1250) — compile-verified, matrix operand order checked by reading, not run on hardware
//
#include <hip/hip_runtime.h>
#include <hip/hip_bf16.h>

typedef __attribute__((ext_vector_type(16))) _Float16 v16h;
typedef __attribute__((ext_vector_type(8)))  _Float16 h8v;
typedef __attribute__((ext_vector_type(8)))  float    v8f;

#define NND 20000
#define NED 40000
#define NGR 800

static __device__ __forceinline__ float sigf(float v){ return 1.0f/(1.0f + __expf(-v)); }

// ---------------------------------------------------------------------------
// Generic tiled WMMA GEMM:  C[M,N] = act(A[M,K] @ B + bias)
// A: f32 row-major (lda).  B: if bnk!=0 -> [N,K] row-major (torch weight),
// else [K,N] row-major.  act: 0 none, 1 relu, 2 lecun_tanh.
// Block = 128 threads (4 waves), 64x64 C tile; wave w owns rows [16w,16w+16).
// Staging: each thread owns 16 contiguous elements of one tile row ->
// 4x global_load_b128 + packed f16 convert + 2x 16B ds_store on the fast path.
// ---------------------------------------------------------------------------
#define GBM 64
#define GBN 64
#define GBK 32

__global__ void __launch_bounds__(128)
k_gemm(const float* __restrict__ A, int lda,
       const float* __restrict__ B, int bnk,
       const float* __restrict__ bias,
       float* __restrict__ C, int ldc,
       int M, int N, int K, int act)
{
    __shared__ __align__(16) _Float16 As[GBM][GBK + 8];   // row stride 80B (16B mult)
    __shared__ __align__(16) _Float16 Bs[GBN][GBK + 8];

    const int tid  = threadIdx.x;
    const int wave = tid >> 5;
    const int lane = tid & 31;
    const int m0   = blockIdx.x * GBM;
    const int n0   = blockIdx.y * GBN;
    const int mrow = wave * 16 + (lane & 15);
    const int g    = lane >> 4;

    const int r_ld  = tid >> 1;          // staged row 0..63
    const int sg_ld = (tid & 1) << 4;    // element offset 0 or 16 within row

    v8f acc[4];
#pragma unroll
    for (int t = 0; t < 4; ++t)
#pragma unroll
        for (int r = 0; r < 8; ++r) acc[t][r] = 0.0f;

    for (int k0 = 0; k0 < K; k0 += GBK) {
        // ---- stage A tile (64 x 32) ----
        {
            int gm = m0 + r_ld;
            if (gm < M && (k0 + GBK) <= K && ((lda & 3) == 0)) {
                const float4* sp = (const float4*)(A + (long long)gm * lda + k0 + sg_ld);
                float tmp[16];
                *(float4*)&tmp[0]  = sp[0];
                *(float4*)&tmp[4]  = sp[1];
                *(float4*)&tmp[8]  = sp[2];
                *(float4*)&tmp[12] = sp[3];
                if (k0 + GBK < K)
                    __builtin_prefetch(A + (long long)gm * lda + k0 + GBK + sg_ld, 0, 1);
                h8v h0, h1;
#pragma unroll
                for (int c = 0; c < 8; ++c) { h0[c] = (_Float16)tmp[c]; h1[c] = (_Float16)tmp[8 + c]; }
                *(h8v*)&As[r_ld][sg_ld]     = h0;
                *(h8v*)&As[r_ld][sg_ld + 8] = h1;
            } else {
#pragma unroll
                for (int c = 0; c < 16; ++c) {
                    int gk = k0 + sg_ld + c;
                    float v = 0.0f;
                    if (gm < M && gk < K) v = A[(long long)gm * lda + gk];
                    As[r_ld][sg_ld + c] = (_Float16)v;
                }
            }
        }
        // ---- stage B tile (64 x 32), stored [n][k] ----
        {
            int gn = n0 + r_ld;
            if (bnk && gn < N && (k0 + GBK) <= K && ((K & 3) == 0)) {
                const float4* sp = (const float4*)(B + (long long)gn * K + k0 + sg_ld);
                float tmp[16];
                *(float4*)&tmp[0]  = sp[0];
                *(float4*)&tmp[4]  = sp[1];
                *(float4*)&tmp[8]  = sp[2];
                *(float4*)&tmp[12] = sp[3];
                h8v h0, h1;
#pragma unroll
                for (int c = 0; c < 8; ++c) { h0[c] = (_Float16)tmp[c]; h1[c] = (_Float16)tmp[8 + c]; }
                *(h8v*)&Bs[r_ld][sg_ld]     = h0;
                *(h8v*)&Bs[r_ld][sg_ld + 8] = h1;
            } else {
#pragma unroll
                for (int c = 0; c < 16; ++c) {
                    int gk = k0 + sg_ld + c;
                    float v = 0.0f;
                    if (gn < N && gk < K)
                        v = bnk ? B[(long long)gn * K + gk] : B[(long long)gk * N + gn];
                    Bs[r_ld][sg_ld + c] = (_Float16)v;
                }
            }
        }
        __syncthreads();

        // A fragment: lane (m,g): halves[0..7]=A[m][g*8+j], halves[8..15]=A[m][16+g*8+j]
        const h8v* arow = (const h8v*)&As[mrow][0];
        h8v alo = arow[g], ahi = arow[2 + g];
        v16h af;
#pragma unroll
        for (int j = 0; j < 8; ++j) { af[j] = alo[j]; af[8 + j] = ahi[j]; }

#pragma unroll
        for (int t = 0; t < 4; ++t) {
            const h8v* brow = (const h8v*)&Bs[t * 16 + (lane & 15)][0];
            h8v blo = brow[g], bhi = brow[2 + g];
            v16h bf;
#pragma unroll
            for (int j = 0; j < 8; ++j) { bf[j] = blo[j]; bf[8 + j] = bhi[j]; }
            acc[t] = __builtin_amdgcn_wmma_f32_16x16x32_f16(
                false, af, false, bf, (short)0, acc[t], false, false);
        }
        __syncthreads();
    }

    const int ncol = lane & 15, gh = lane >> 4;
    for (int t = 0; t < 4; ++t) {
        int gn = n0 + t * 16 + ncol;
        if (gn >= N) continue;
        float bv = bias ? bias[gn] : 0.0f;
#pragma unroll
        for (int r = 0; r < 8; ++r) {
            int gm = m0 + wave * 16 + r + 8 * gh;
            if (gm >= M) continue;
            float v = acc[t][r] + bv;
            if (act == 1)      v = fmaxf(v, 0.0f);
            else if (act == 2) v = 1.7159f * tanhf(0.666f * v);
            C[(long long)gm * ldc + gn] = v;
        }
    }
}

// ---------------------------------------------------------------------------
// Fused NNConv message: msg[e,o] = x[src_e] @ reshape(h_e@w2^T+b2, cin,cout)
// Implicit GEMM: A[e, i*32+kk] = x[src_e,i]*h[e,kk] built in registers from
// LDS tiles; B[(i,kk), o] = w2[i*cout+o, kk].  b2 folded as extra K-steps.
// cin<=64, cout in {32,64}. Block = 128 threads, 64 edges per block.
// ---------------------------------------------------------------------------
__global__ void __launch_bounds__(128)
k_nnconv_msg(const float* __restrict__ x, int xld, int cin,
             const int* __restrict__ src,
             const float* __restrict__ h,      // E x 32
             const float* __restrict__ w2,     // (cin*cout) x 32
             const float* __restrict__ b2,     // cin*cout
             float* __restrict__ msg, int E, int cout)
{
    __shared__ __align__(16) float hs[64][36];            // stride 144B
    __shared__ __align__(16) float xs[64][68];            // stride 272B
    __shared__ __align__(16) _Float16 Bs[64][40];         // stride 80B

    const int tid  = threadIdx.x;
    const int wave = tid >> 5;
    const int lane = tid & 31;
    const int e0   = blockIdx.x * 64;
    const int mrow = wave * 16 + (lane & 15);
    const int g    = lane >> 4;
    const int ntile = cout >> 4;

    v8f acc[4];
#pragma unroll
    for (int t = 0; t < 4; ++t)
#pragma unroll
        for (int r = 0; r < 8; ++r) acc[t][r] = 0.0f;

    // stage h tile: 64 x 32 f32, each thread 16 contiguous
    {
        int r = tid >> 1, sgm = (tid & 1) << 4;
        int e = e0 + r;
        if (e < E) {
            const float4* sp = (const float4*)(h + (long long)e * 32 + sgm);
#pragma unroll
            for (int u = 0; u < 4; ++u) *(float4*)&hs[r][sgm + 4 * u] = sp[u];
        } else {
#pragma unroll
            for (int c = 0; c < 16; ++c) hs[r][sgm + c] = 0.0f;
        }
    }
    // stage gathered x[src] tile: 64 x 64 f32 (zero-padded past cin)
    {
        int r = tid >> 1, cb = (tid & 1) << 5;   // 0 or 32
        int e = e0 + r;
        if (e < E && ((xld & 3) == 0) && (cb + 32) <= cin) {
            const float4* sp = (const float4*)(x + (long long)src[e] * xld + cb);
#pragma unroll
            for (int u = 0; u < 8; ++u) *(float4*)&xs[r][cb + 4 * u] = sp[u];
        } else {
            long long base = (e < E) ? (long long)src[e] * xld : 0;
#pragma unroll
            for (int c = 0; c < 32; ++c) {
                int cc = cb + c;
                float v = 0.0f;
                if (e < E && cc < cin) v = x[base + cc];
                xs[r][cc] = v;
            }
        }
    }
    __syncthreads();

    // main contraction: K-step per input channel i (32 kk values each)
    for (int i = 0; i < cin; ++i) {
        // stage B slice: w2 rows [i*cout .. i*cout+cout), each 32 contiguous f32
        {
            int base = tid * 16;
            if (base < cout * 32) {
                int n = base >> 5, kk = base & 31;   // kk = 0 or 16
                const float4* sp = (const float4*)(w2 + ((long long)i * cout + n) * 32 + kk);
                float tmp[16];
                *(float4*)&tmp[0]  = sp[0];
                *(float4*)&tmp[4]  = sp[1];
                *(float4*)&tmp[8]  = sp[2];
                *(float4*)&tmp[12] = sp[3];
                h8v h0, h1;
#pragma unroll
                for (int c = 0; c < 8; ++c) { h0[c] = (_Float16)tmp[c]; h1[c] = (_Float16)tmp[8 + c]; }
                *(h8v*)&Bs[n][kk]     = h0;
                *(h8v*)&Bs[n][kk + 8] = h1;
            }
        }
        __syncthreads();

        float xv = xs[mrow][i];
        v16h af;
#pragma unroll
        for (int j = 0; j < 8; ++j) {
            af[j]     = (_Float16)(xv * hs[mrow][g * 8 + j]);
            af[8 + j] = (_Float16)(xv * hs[mrow][16 + g * 8 + j]);
        }
#pragma unroll
        for (int t = 0; t < 4; ++t) {
            if (t < ntile) {
                const h8v* brow = (const h8v*)&Bs[t * 16 + (lane & 15)][0];
                h8v blo = brow[g], bhi = brow[2 + g];
                v16h bf;
#pragma unroll
                for (int j = 0; j < 8; ++j) { bf[j] = blo[j]; bf[8 + j] = bhi[j]; }
                acc[t] = __builtin_amdgcn_wmma_f32_16x16x32_f16(
                    false, af, false, bf, (short)0, acc[t], false, false);
            }
        }
        __syncthreads();
    }

    // b2 term: msg += x[src] @ b2mat  (b2 viewed as cin x cout, strided -> scalar)
    for (int t2 = 0; t2 * 32 < cin; ++t2) {
        for (int idx = tid; idx < cout * 32; idx += 128) {
            int n = idx >> 5, kk = idx & 31;
            int ii = t2 * 32 + kk;
            Bs[n][kk] = (ii < cin) ? (_Float16)b2[(long long)ii * cout + n]
                                   : (_Float16)0.0f;
        }
        __syncthreads();
        v16h af;
#pragma unroll
        for (int j = 0; j < 8; ++j) {
            af[j]     = (_Float16)xs[mrow][t2 * 32 + g * 8 + j];
            af[8 + j] = (_Float16)xs[mrow][t2 * 32 + 16 + g * 8 + j];
        }
#pragma unroll
        for (int t = 0; t < 4; ++t) {
            if (t < ntile) {
                const h8v* brow = (const h8v*)&Bs[t * 16 + (lane & 15)][0];
                h8v blo = brow[g], bhi = brow[2 + g];
                v16h bf;
#pragma unroll
                for (int j = 0; j < 8; ++j) { bf[j] = blo[j]; bf[8 + j] = bhi[j]; }
                acc[t] = __builtin_amdgcn_wmma_f32_16x16x32_f16(
                    false, af, false, bf, (short)0, acc[t], false, false);
            }
        }
        __syncthreads();
    }

    const int ncol = lane & 15, gh = lane >> 4;
    for (int t = 0; t < 4; ++t) {
        if (t >= ntile) break;
        int n = t * 16 + ncol;
#pragma unroll
        for (int r = 0; r < 8; ++r) {
            int e = e0 + wave * 16 + r + 8 * gh;
            if (e < E) msg[(long long)e * cout + n] = acc[t][r];
        }
    }
}

// ---------------------------------------------------------------------------
// small kernels
// ---------------------------------------------------------------------------
__global__ void k_fill(float* p, float v, long long n){
    long long i = (long long)blockIdx.x * blockDim.x + threadIdx.x;
    if (i < n) p[i] = v;
}
__global__ void k_fillu(unsigned* p, unsigned v, long long n){
    long long i = (long long)blockIdx.x * blockDim.x + threadIdx.x;
    if (i < n) p[i] = v;
}
__global__ void k_deg(const int* dst, float* cnt, int e){
    int i = blockIdx.x * blockDim.x + threadIdx.x;
    if (i < e) atomicAdd(&cnt[dst[i]], 1.0f);
}
__global__ void k_scatter(const float* msg, const int* dst, float* agg, long long e, int c){
    long long i = (long long)blockIdx.x * blockDim.x + threadIdx.x;
    if (i < e * c) {
        int ed = (int)(i / c), ch = (int)(i - (long long)ed * c);
        atomicAdd(&agg[(long long)dst[ed] * c + ch], msg[i]);
    }
}
__global__ void k_fin(const float* agg, const float* cnt, const float* xr,
                      float* out, int n, int c, int act){
    long long i = (long long)blockIdx.x * blockDim.x + threadIdx.x;
    if (i < (long long)n * c) {
        int nd = (int)(i / c);
        float v = agg[i] / fmaxf(cnt[nd], 1.0f) + xr[i];
        if (act) v = fmaxf(v, 0.0f);
        out[i] = v;
    }
}
__global__ void k_copy_cols(float* dst, int dld, int doff,
                            const float* src, int sld, int soff,
                            int rows, int cols){
    long long i = (long long)blockIdx.x * blockDim.x + threadIdx.x;
    if (i < (long long)rows * cols) {
        int r = (int)(i / cols), c = (int)(i - (long long)r * cols);
        dst[(long long)r * dld + doff + c] = src[(long long)r * sld + soff + c];
    }
}
// CfC cell 1: x2 = x[:,42:48]; 5 unrolled steps of an 18->12 gated cell
__global__ void k_cfc1(const float* __restrict__ x, int xld, int xoff,
                       const float* w1, const float* b1,
                       const float* w2, const float* b2,
                       const float* wa, const float* ba,
                       const float* wb, const float* bb,
                       float* out30, int n)
{
    int nid = blockIdx.x * blockDim.x + threadIdx.x;
    if (nid >= n) return;
    float x2[6], h[12];
    for (int j = 0; j < 6; ++j) { x2[j] = x[(long long)nid * xld + xoff + j]; }
    for (int j = 0; j < 6; ++j) { h[j] = x2[j]; h[6 + j] = x2[j]; }
    for (int s = 0; s < 5; ++s) {
        float xh[18];
        for (int j = 0; j < 6;  ++j) xh[j]     = x2[j];
        for (int j = 0; j < 12; ++j) xh[6 + j] = h[j];
        for (int o = 0; o < 12; ++o) {
            float s1 = b1[o], s2 = b2[o], st = ba[o] + bb[o];
            for (int i = 0; i < 18; ++i) {
                float xi = xh[i];
                s1 += w1[o * 18 + i] * xi;
                s2 += w2[o * 18 + i] * xi;
                st += (wa[o * 18 + i] + wb[o * 18 + i]) * xi;
            }
            float ti = sigf(st);
            h[o] = tanhf(s1) * (1.0f - ti) + ti * tanhf(s2);
        }
        for (int j = 0; j < 6; ++j) out30[(long long)nid * 30 + s * 6 + j] = h[6 + j];
    }
}
// 2-token, 4-head (dh=8) attention
__global__ void k_attn(const float* q, int qs, const float* k, int ks,
                       const float* v, int vs, float* out, int n)
{
    int t = blockIdx.x * blockDim.x + threadIdx.x;
    if (t >= n * 4) return;
    int nd = t >> 2, hd = t & 3;
    float qv[2][8], kv[2][8], vv[2][8];
    for (int s = 0; s < 2; ++s) {
        long long row = (long long)s * n + nd;
        for (int d = 0; d < 8; ++d) {
            qv[s][d] = q[row * qs + hd * 8 + d];
            kv[s][d] = k[row * ks + hd * 8 + d];
            vv[s][d] = v[row * vs + hd * 8 + d];
        }
    }
    const float sc = 0.35355339059327f; // 1/sqrt(8)
    for (int qq = 0; qq < 2; ++qq) {
        float s0 = 0.0f, s1 = 0.0f;
        for (int d = 0; d < 8; ++d) { s0 += qv[qq][d] * kv[0][d]; s1 += qv[qq][d] * kv[1][d]; }
        s0 *= sc; s1 *= sc;
        float m = fmaxf(s0, s1);
        float a0 = __expf(s0 - m), a1 = __expf(s1 - m);
        float den = a0 + a1; a0 /= den; a1 /= den;
        long long row = (long long)qq * n + nd;
        for (int d = 0; d < 8; ++d)
            out[row * 32 + hd * 8 + d] = a0 * vv[0][d] + a1 * vv[1][d];
    }
}
// y = LN(a + b) * g + bt   (d = 32)
__global__ void k_addln(const float* a, const float* b,
                        const float* g, const float* bt, float* out, int rows)
{
    int r = blockIdx.x * blockDim.x + threadIdx.x;
    if (r >= rows) return;
    float t[32], m = 0.0f;
    for (int i = 0; i < 32; ++i) { t[i] = a[(long long)r * 32 + i] + b[(long long)r * 32 + i]; m += t[i]; }
    m *= (1.0f / 32.0f);
    float var = 0.0f;
    for (int i = 0; i < 32; ++i) { float d = t[i] - m; var += d * d; }
    var *= (1.0f / 32.0f);
    float inv = rsqrtf(var + 1e-5f);
    for (int i = 0; i < 32; ++i)
        out[(long long)r * 32 + i] = (t[i] - m) * inv * g[i] + bt[i];
}
__global__ void k_seqmean(const float* mem, float* dec, int n){
    long long i = (long long)blockIdx.x * blockDim.x + threadIdx.x;
    if (i < (long long)n * 32) {
        float v = 0.5f * (mem[i] + mem[(long long)n * 32 + i]);
        dec[i] = v; dec[(long long)n * 32 + i] = v;
    }
}
__global__ void k_trmean(const float* dec, float* tr, int n){
    long long i = (long long)blockIdx.x * blockDim.x + threadIdx.x;
    if (i < (long long)n * 32)
        tr[i] = fmaxf(0.5f * (dec[i] + dec[(long long)n * 32 + i]), 0.0f);
}
__global__ void k_cfc2comb(const float* f1, const float* f2, const float* ta,
                           const float* tb, float* xm, long long n){
    long long i = (long long)blockIdx.x * blockDim.x + threadIdx.x;
    if (i < n) {
        float ti = sigf(ta[i] + tb[i]);
        float v = tanhf(f1[i]) * (1.0f - ti) + ti * tanhf(f2[i]);
        xm[i] = fmaxf(v, 0.0f);
    }
}
// set2set pieces
__global__ void k_lstm(const float* g1, const float* g2, float* c, float* h, int G){
    int t = blockIdx.x * blockDim.x + threadIdx.x;
    if (t >= G * 64) return;
    int g = t >> 6, j = t & 63;
    float ii = g1[g * 256 + j]       + g2[g * 256 + j];
    float ff = g1[g * 256 + 64 + j]  + g2[g * 256 + 64 + j];
    float gg = g1[g * 256 + 128 + j] + g2[g * 256 + 128 + j];
    float oo = g1[g * 256 + 192 + j] + g2[g * 256 + 192 + j];
    float cc = sigf(ff) * c[t] + sigf(ii) * tanhf(gg);
    c[t] = cc;
    h[t] = sigf(oo) * tanhf(cc);
}
__global__ void k_s2s_e(const float* xs, const int* batch, const float* h, float* e, int n){
    int i = blockIdx.x * blockDim.x + threadIdx.x;
    if (i >= n) return;
    const float* hp = &h[(long long)batch[i] * 64];
    float s = 0.0f;
    for (int j = 0; j < 64; ++j) s += xs[(long long)i * 64 + j] * hp[j];
    e[i] = s;
}
static __device__ __forceinline__ unsigned fkey(float f){
    unsigned u = __float_as_uint(f);
    return (u & 0x80000000u) ? ~u : (u | 0x80000000u);
}
static __device__ __forceinline__ float funkey(unsigned k){
    unsigned u = (k & 0x80000000u) ? (k & 0x7fffffffu) : ~k;
    return __uint_as_float(u);
}
__global__ void k_s2s_max(const float* e, const int* batch, unsigned* emax, int n){
    int i = blockIdx.x * blockDim.x + threadIdx.x;
    if (i < n) atomicMax(&emax[batch[i]], fkey(e[i]));
}
__global__ void k_s2s_a(const float* e, const int* batch, const unsigned* emax,
                        float* a, float* den, int n){
    int i = blockIdx.x * blockDim.x + threadIdx.x;
    if (i >= n) return;
    int b = batch[i];
    float av = __expf(e[i] - funkey(emax[b]));
    a[i] = av;
    atomicAdd(&den[b], av);
}
__global__ void k_s2s_r(const float* a, const float* den, const int* batch,
                        const float* xs, float* r, int n){
    int i = blockIdx.x * blockDim.x + threadIdx.x;
    if (i >= n) return;
    int b = batch[i];
    float w = a[i] / den[b];
    for (int j = 0; j < 64; ++j)
        atomicAdd(&r[(long long)b * 64 + j], w * xs[(long long)i * 64 + j]);
}
__global__ void k_qstar(const float* h, const float* r, float* qs, int G){
    int t = blockIdx.x * blockDim.x + threadIdx.x;
    if (t >= G * 128) return;
    int g = t >> 7, j = t & 127;
    qs[t] = (j < 64) ? h[g * 64 + j] : r[g * 64 + (j - 64)];
}

// ---------------------------------------------------------------------------
extern "C" void kernel_launch(void* const* d_in, const int* in_sizes, int n_in,
                              void* d_out, int out_size, void* d_ws, size_t ws_size,
                              hipStream_t stream)
{
    (void)in_sizes; (void)n_in; (void)out_size; (void)ws_size;
    const float* X     = (const float*)d_in[0];
    const int*   EIDX  = (const int*)d_in[1];
    const float* EATTR = (const float*)d_in[2];
    const int*   BATCH = (const int*)d_in[3];
    const int*   SRC   = EIDX;
    const int*   DST   = EIDX + NED;
    auto P = [&](int i){ return (const float*)d_in[4 + i]; };
    // param index map (relative to first param):
    // a11=0..5 a21=6..11 sg1=12..17 sg2=18..23 (w1,b1,w2,b2,root,bias)
    // cfc1=24..31  cfc2_bb=32,33  cfc2=34..41  x22=42,43  last=44,45
    // enc0=46..57 enc1=58..69  dec0=70..87 dec1=88..105
    // s2s=106..109  fc=110..117

    float* W = (float*)d_ws;
    size_t off = 0;
    auto alloc = [&](size_t n){ float* p = W + off; off += (n + 255) & ~(size_t)255; return p; };

    float* cnt    = alloc(NND);
    float* hE     = alloc((size_t)NED * 32);
    float* msg    = alloc((size_t)NED * 64);
    float* agg    = alloc((size_t)NND * 64);
    float* xr     = alloc((size_t)NND * 64);
    float* x1     = alloc((size_t)NND * 32);
    float* x2raw  = alloc((size_t)NND * 30);
    float* x2a    = alloc((size_t)NND * 32);
    float* x2b    = alloc((size_t)NND * 32);
    float* mem    = alloc((size_t)2 * NND * 32);
    float* qkv    = alloc((size_t)2 * NND * 96);
    float* atto   = alloc((size_t)2 * NND * 32);
    float* projo  = alloc((size_t)2 * NND * 32);
    float* ffh    = alloc((size_t)2 * NND * 64);
    float* dec    = alloc((size_t)2 * NND * 32);
    float* trb    = alloc((size_t)NND * 32);
    float* cat128 = alloc((size_t)NND * 128);
    float* bb     = alloc((size_t)NND * 128);
    float* t64a   = alloc((size_t)NND * 64);
    float* t64b   = alloc((size_t)NND * 64);
    float* t64c   = alloc((size_t)NND * 64);
    float* t64d   = alloc((size_t)NND * 64);
    float* xm     = alloc((size_t)NND * 64);
    float* xm3    = alloc((size_t)NND * 64);
    float* sgA    = alloc((size_t)NND * 64);
    float* sgB    = alloc((size_t)NND * 64);
    float* gates1 = alloc((size_t)NGR * 256);
    float* gates2 = alloc((size_t)NGR * 256);
    float* qstar  = alloc((size_t)NGR * 128);
    float* hG     = alloc((size_t)NGR * 64);
    float* cG     = alloc((size_t)NGR * 64);
    float* rG     = alloc((size_t)NGR * 64);
    float* eN     = alloc(NND);
    float* aN     = alloc(NND);
    float* denG   = alloc(NGR);
    unsigned* emaxG = (unsigned*)alloc(NGR);
    float* fco1   = alloc((size_t)NGR * 512);
    float* fco2   = alloc((size_t)NGR * 256);
    float* fco3   = alloc((size_t)NGR * 64);

    auto cdiv = [](long long a, long long b){ return (int)((a + b - 1) / b); };
    auto gemm = [&](const float* A, int lda, const float* B, int bnk, const float* bias,
                    float* C, int ldc, int M, int N, int K, int act){
        dim3 grid(cdiv(M, GBM), cdiv(N, GBN));
        k_gemm<<<grid, 128, 0, stream>>>(A, lda, B, bnk, bias, C, ldc, M, N, K, act);
    };
    auto fill = [&](float* p, float v, long long n){
        k_fill<<<cdiv(n, 256), 256, 0, stream>>>(p, v, n);
    };

    // degree counts (shared by all nnconvs)
    fill(cnt, 0.0f, NND);
    k_deg<<<cdiv(NED, 256), 256, 0, stream>>>(DST, cnt, NED);

    // full NNConv layer
    auto nnconv = [&](const float* xin, int xld, int cin, int cout, int pb,
                      float* out, int act){
        gemm(EATTR, 10, P(pb + 0), 1, P(pb + 1), hE, 32, NED, 32, 10, 1); // h = relu(ea@w1^T+b1)
        k_nnconv_msg<<<cdiv(NED, 64), 128, 0, stream>>>(
            xin, xld, cin, SRC, hE, P(pb + 2), P(pb + 3), msg, NED, cout);
        fill(agg, 0.0f, (long long)NND * cout);
        k_scatter<<<cdiv((long long)NED * cout, 256), 256, 0, stream>>>(msg, DST, agg, NED, cout);
        gemm(xin, xld, P(pb + 4), 0, P(pb + 5), xr, cout, NND, cout, cin, 0); // x@root + bias
        k_fin<<<cdiv((long long)NND * cout, 256), 256, 0, stream>>>(agg, cnt, xr, out, NND, cout, act);
    };

    // ---- branch 1: x1 = relu(nnconv_a11(x[:,:42])) ----
    nnconv(X, 48, 42, 32, 0, x1, 1);

    // ---- branch 2: CfC1 -> x22 -> nnconv_a21 ----
    k_cfc1<<<cdiv(NND, 256), 256, 0, stream>>>(X, 48, 42,
        P(24), P(25), P(26), P(27), P(28), P(29), P(30), P(31), x2raw, NND);
    gemm(x2raw, 30, P(42), 1, P(43), x2a, 32, NND, 32, 30, 1);
    nnconv(x2a, 32, 32, 32, 6, x2b, 1);

    // ---- transformer: mem = [x1; x2b] ----
    (void)hipMemcpyAsync(mem, x1, (size_t)NND * 32 * 4, hipMemcpyDeviceToDevice, stream);
    (void)hipMemcpyAsync(mem + (size_t)NND * 32, x2b, (size_t)NND * 32 * 4,
                         hipMemcpyDeviceToDevice, stream);

    auto mha_self = [&](float* io, int inw, int inb, int ow, int ob, int ng, int nbt){
        gemm(io, 32, P(inw), 1, P(inb), qkv, 96, 2 * NND, 96, 32, 0);
        k_attn<<<cdiv((long long)NND * 4, 256), 256, 0, stream>>>(
            qkv, 96, qkv + 32, 96, qkv + 64, 96, atto, NND);
        gemm(atto, 32, P(ow), 1, P(ob), projo, 32, 2 * NND, 32, 32, 0);
        k_addln<<<cdiv(2 * NND, 256), 256, 0, stream>>>(io, projo, P(ng), P(nbt), io, 2 * NND);
    };
    auto mha_cross = [&](float* io, const float* kvsrc, int inw, int inb,
                         int ow, int ob, int ng, int nbt){
        gemm(io, 32, P(inw), 1, P(inb), qkv, 96, 2 * NND, 32, 32, 0);            // q
        gemm(kvsrc, 32, P(inw) + 32 * 32, 1, P(inb) + 32, qkv + 32, 96,
             2 * NND, 64, 32, 0);                                                // k,v
        k_attn<<<cdiv((long long)NND * 4, 256), 256, 0, stream>>>(
            qkv, 96, qkv + 32, 96, qkv + 64, 96, atto, NND);
        gemm(atto, 32, P(ow), 1, P(ob), projo, 32, 2 * NND, 32, 32, 0);
        k_addln<<<cdiv(2 * NND, 256), 256, 0, stream>>>(io, projo, P(ng), P(nbt), io, 2 * NND);
    };
    auto ffn = [&](float* io, int l1w, int l1b, int l2w, int l2b, int ng, int nbt){
        gemm(io, 32, P(l1w), 1, P(l1b), ffh, 64, 2 * NND, 64, 32, 1);
        gemm(ffh, 64, P(l2w), 1, P(l2b), projo, 32, 2 * NND, 32, 64, 0);
        k_addln<<<cdiv(2 * NND, 256), 256, 0, stream>>>(io, projo, P(ng), P(nbt), io, 2 * NND);
    };

    for (int l = 0; l < 2; ++l) {
        int b = 46 + l * 12;
        mha_self(mem, b + 0, b + 1, b + 2, b + 3, b + 8, b + 9);
        ffn(mem, b + 4, b + 5, b + 6, b + 7, b + 10, b + 11);
    }
    k_seqmean<<<cdiv((long long)NND * 32, 256), 256, 0, stream>>>(mem, dec, NND);
    for (int l = 0; l < 2; ++l) {
        int b = 70 + l * 18;
        mha_self(dec, b + 0, b + 1, b + 2, b + 3, b + 12, b + 13);
        mha_cross(dec, mem, b + 4, b + 5, b + 6, b + 7, b + 14, b + 15);
        ffn(dec, b + 8, b + 9, b + 10, b + 11, b + 16, b + 17);
    }
    k_trmean<<<cdiv((long long)NND * 32, 256), 256, 0, stream>>>(dec, trb, NND);

    // ---- cfc2: bb_in = [x1, x2b, tr, tr] ----
    k_copy_cols<<<cdiv((long long)NND * 32, 256), 256, 0, stream>>>(cat128, 128, 0,  x1,  32, 0, NND, 32);
    k_copy_cols<<<cdiv((long long)NND * 32, 256), 256, 0, stream>>>(cat128, 128, 32, x2b, 32, 0, NND, 32);
    k_copy_cols<<<cdiv((long long)NND * 32, 256), 256, 0, stream>>>(cat128, 128, 64, trb, 32, 0, NND, 32);
    k_copy_cols<<<cdiv((long long)NND * 32, 256), 256, 0, stream>>>(cat128, 128, 96, trb, 32, 0, NND, 32);
    gemm(cat128, 128, P(32), 1, P(33), bb, 128, NND, 128, 128, 2);   // lecun_tanh
    gemm(bb, 128, P(34), 1, P(35), t64a, 64, NND, 64, 128, 0);
    gemm(bb, 128, P(36), 1, P(37), t64b, 64, NND, 64, 128, 0);
    gemm(bb, 128, P(38), 1, P(39), t64c, 64, NND, 64, 128, 0);
    gemm(bb, 128, P(40), 1, P(41), t64d, 64, NND, 64, 128, 0);
    k_cfc2comb<<<cdiv((long long)NND * 64, 256), 256, 0, stream>>>(t64a, t64b, t64c, t64d, xm,
                                                                   (long long)NND * 64);
    // xm2 = [xm, x1, x2b] ; xm3 = relu(xm2 @ last)
    k_copy_cols<<<cdiv((long long)NND * 64, 256), 256, 0, stream>>>(cat128, 128, 0,  xm,  64, 0, NND, 64);
    k_copy_cols<<<cdiv((long long)NND * 32, 256), 256, 0, stream>>>(cat128, 128, 64, x1,  32, 0, NND, 32);
    k_copy_cols<<<cdiv((long long)NND * 32, 256), 256, 0, stream>>>(cat128, 128, 96, x2b, 32, 0, NND, 32);
    gemm(cat128, 128, P(44), 1, P(45), xm3, 64, NND, 64, 128, 1);

    // ---- sg chain ----
    nnconv(xm3, 64, 64, 64, 12, sgA, 1);     // sg1, relu
    nnconv(sgA, 64, 64, 64, 18, sgB, 1);     // sg2, relu
    nnconv(sgB, 64, 64, 64, 18, sgA, 0);     // sg2, no act

    // ---- set2set (2 steps) ----
    fill(qstar, 0.0f, (long long)NGR * 128);
    fill(hG, 0.0f, (long long)NGR * 64);
    fill(cG, 0.0f, (long long)NGR * 64);
    for (int s = 0; s < 2; ++s) {
        gemm(qstar, 128, P(106), 1, P(107), gates1, 256, NGR, 256, 128, 0);
        gemm(hG, 64,    P(108), 1, P(109), gates2, 256, NGR, 256, 64, 0);
        k_lstm<<<cdiv((long long)NGR * 64, 256), 256, 0, stream>>>(gates1, gates2, cG, hG, NGR);
        k_s2s_e<<<cdiv(NND, 256), 256, 0, stream>>>(sgA, BATCH, hG, eN, NND);
        k_fillu<<<cdiv(NGR, 256), 256, 0, stream>>>(emaxG, 0x007FFFFFu, NGR); // key(-inf)
        fill(denG, 0.0f, NGR);
        fill(rG, 0.0f, (long long)NGR * 64);
        k_s2s_max<<<cdiv(NND, 256), 256, 0, stream>>>(eN, BATCH, emaxG, NND);
        k_s2s_a<<<cdiv(NND, 256), 256, 0, stream>>>(eN, BATCH, emaxG, aN, denG, NND);
        k_s2s_r<<<cdiv(NND, 256), 256, 0, stream>>>(aN, denG, BATCH, sgA, rG, NND);
        k_qstar<<<cdiv((long long)NGR * 128, 256), 256, 0, stream>>>(hG, rG, qstar, NGR);
    }

    // ---- fc head ----
    gemm(qstar, 128, P(110), 1, P(111), fco1, 512, NGR, 512, 128, 1);
    gemm(fco1, 512,  P(112), 1, P(113), fco2, 256, NGR, 256, 512, 1);
    gemm(fco2, 256,  P(114), 1, P(115), fco3, 64,  NGR, 64,  256, 1);
    gemm(fco3, 64,   P(116), 1, P(117), (float*)d_out, 1, NGR, 1, 64, 0);

    // ---- outputs: (o, x1, x2_outputs, tr) ----
    float* OUT = (float*)d_out;
    (void)hipMemcpyAsync(OUT + 800, x1, (size_t)NND * 32 * 4, hipMemcpyDeviceToDevice, stream);
    (void)hipMemcpyAsync(OUT + 800 + (size_t)NND * 32, x2b, (size_t)NND * 32 * 4,
                         hipMemcpyDeviceToDevice, stream);
    (void)hipMemcpyAsync(OUT + 800 + (size_t)2 * NND * 32, trb, (size_t)NND * 32 * 4,
                         hipMemcpyDeviceToDevice, stream);
}
